// CaGCN_26714696581624
// MI455X (gfx1250) — compile-verified, matrix operand order checked by reading, not run on hardware
//
#include <hip/hip_runtime.h>
#include <hip/hip_bf16.h>
#include <math.h>

// ---------------------------------------------------------------------------
// CDNA5 (gfx1250) CaGCN pipeline.
// Dense linear layers use v_wmma_f32_16x16x32_f16 (wave32 WMMA) with
// pre-packed f16 B fragments (contiguous b128 loads) and float4 A loads.
// Sparse neighbor aggregation uses global_atomic_add_f32 (L2-resident tables).
// ---------------------------------------------------------------------------

typedef __attribute__((ext_vector_type(16))) _Float16 v16h;
typedef __attribute__((ext_vector_type(8)))  float    v8f;

static __device__ __forceinline__ int gtid() {
    return blockIdx.x * blockDim.x + threadIdx.x;
}

// ---------------- degree / normalization ----------------

__global__ void k_init_deg(float* __restrict__ deg, int n) {
    int i = gtid();
    if (i < n) deg[i] = 1.0f;   // self loop contributes 1
}

__global__ void k_accum_deg(const int* __restrict__ dst, int e,
                            float* __restrict__ deg) {
    int i = gtid();
    if (i < e) atomicAdd(&deg[dst[i]], 1.0f);
}

__global__ void k_dinv(const float* __restrict__ deg,
                       float* __restrict__ dinv, int n) {
    int i = gtid();
    if (i < n) dinv[i] = rsqrtf(fmaxf(deg[i], 1.0f));
}

// ---------------- B packing for WMMA ----------------
// Pack fp32 B[K x (16*NT)] (row-major) into f16 fragments stored in the exact
// per-lane order the 16-bit B operand wants, so each lane's 16 halves are one
// contiguous 32-byte load. Fragment index = (k_chunk*NT + t)*32 + lane.

template <int NT>
__global__ void k_pack_b(const float* __restrict__ B,
                         _Float16* __restrict__ Bp, int K) {
    const int N = NT * 16;
    const int chunks = K >> 5;
    int i = gtid();
    if (i >= chunks * NT * 32) return;
    const int lane = i & 31;
    const int t    = (i >> 5) % NT;
    const int c    = (i >> 5) / NT;
    const int r    = lane & 15;
    const int kh   = (lane >> 4) << 3;        // 0 or 8
    const int k0   = c << 5;
    const int col  = t * 16 + r;
    _Float16* o = Bp + (size_t)i * 16;
#pragma unroll
    for (int j = 0; j < 8; ++j) {
        o[j]     = (_Float16)B[(size_t)(k0 + kh + j) * N + col];
        o[8 + j] = (_Float16)B[(size_t)(k0 + 16 + kh + j) * N + col];
    }
}

// ---------------- WMMA GEMM: C[M x (16*NT)] = A[M x K] * Bp ----------------
// One wave per 16-row M tile, NT column tiles. fp32 accumulate.
// A lane layout (16-bit A 16x32): lanes 0-15 -> M=lane, K = {k0+kh..+7,
// k0+16+kh..+7}, kh = (lane>=16)*8. D layout: VGPR j holds
// (M = j + 8*(lane>=16), N = lane&15).

template <int NT>
__global__ void k_wmma_gemm(const float* __restrict__ A,
                            const _Float16* __restrict__ Bp,
                            float* __restrict__ C,
                            int M, int K) {
    const int N    = NT * 16;
    const int wave = threadIdx.x >> 5;
    const int lane = threadIdx.x & 31;
    const int tile = blockIdx.x * (blockDim.x >> 5) + wave;
    if (tile * 16 >= M) return;          // wave-uniform: EXEC stays all-ones

    const int r   = lane & 15;
    const int kh  = (lane >> 4) << 3;    // 0 or 8
    const int row = tile * 16 + r;
    const float* __restrict__ arow = A + (size_t)row * K;
    const v16h* __restrict__ bfrag = (const v16h*)Bp;

    v8f zero = {};
    v8f acc[NT];
#pragma unroll
    for (int t = 0; t < NT; ++t) acc[t] = zero;

    for (int k0 = 0; k0 < K; k0 += 32) {
        const float4* __restrict__ a0 =
            reinterpret_cast<const float4*>(arow + k0 + kh);
        const float4* __restrict__ a1 =
            reinterpret_cast<const float4*>(arow + k0 + 16 + kh);
        float4 p0 = a0[0], p1 = a0[1];
        float4 q0 = a1[0], q1 = a1[1];
        v16h a;
        a[0]  = (_Float16)p0.x; a[1]  = (_Float16)p0.y;
        a[2]  = (_Float16)p0.z; a[3]  = (_Float16)p0.w;
        a[4]  = (_Float16)p1.x; a[5]  = (_Float16)p1.y;
        a[6]  = (_Float16)p1.z; a[7]  = (_Float16)p1.w;
        a[8]  = (_Float16)q0.x; a[9]  = (_Float16)q0.y;
        a[10] = (_Float16)q0.z; a[11] = (_Float16)q0.w;
        a[12] = (_Float16)q1.x; a[13] = (_Float16)q1.y;
        a[14] = (_Float16)q1.z; a[15] = (_Float16)q1.w;

        const int cidx = (k0 >> 5) * NT;
#pragma unroll
        for (int t = 0; t < NT; ++t) {
            v16h b = bfrag[(size_t)(cidx + t) * 32 + lane];
            acc[t] = __builtin_amdgcn_wmma_f32_16x16x32_f16(
                false, a, false, b, (short)0, acc[t], false, false);
        }
    }

    const int mbase = (lane < 16) ? 0 : 8;
    const int ncol  = lane & 15;
#pragma unroll
    for (int j = 0; j < 8; ++j) {
        const size_t orow = (size_t)(tile * 16 + mbase + j);
#pragma unroll
        for (int t = 0; t < NT; ++t)
            C[orow * N + t * 16 + ncol] = acc[t][j];
    }
}

// ---------------- aggregation ----------------

// agg[i*C+c] = bias[c] + hw[i*C+c] * dinv[i]^2   (bias + self-loop term)
template <int C>
__global__ void k_init_agg(const float* __restrict__ hw,
                           const float* __restrict__ dinv,
                           const float* __restrict__ bias,
                           float* __restrict__ agg, int n) {
    int i = gtid();
    if (i >= n * C) return;
    int node = i / C, c = i % C;     // C is a power of two -> shift/mask
    float di = dinv[node];
    agg[i] = bias[c] + hw[i] * di * di;
}

// agg[dst*C+c] += hw[src*C+c] * dinv[src]*dinv[dst]
template <int C>
__global__ void k_scatter(const float* __restrict__ hw,
                          const int* __restrict__ src,
                          const int* __restrict__ dst,
                          const float* __restrict__ dinv,
                          float* __restrict__ agg, int e) {
    int i = gtid();
    int edge = i / C, c = i % C;     // compile-time C -> shift/mask
    if (edge >= e) return;
    if (edge + 4096 < e) {           // gfx1250 global_prefetch path
        __builtin_prefetch(src + edge + 4096, 0, 0);
        __builtin_prefetch(dst + edge + 4096, 0, 0);
    }
    int s = src[edge], d = dst[edge];
    float w = dinv[s] * dinv[d];
    atomicAdd(&agg[(size_t)d * C + c], hw[(size_t)s * C + c] * w);
}

// tlin[i] = dot(relu(h[i, 0:16]), w[0:16])   (ReLU folded into the read)
__global__ void k_matvec16_relu(const float* __restrict__ h,
                                const float* __restrict__ w,
                                float* __restrict__ out, int n) {
    int i = gtid();
    if (i >= n) return;
    const float4* hr = (const float4*)(h + (size_t)i * 16);
    float s = 0.0f;
#pragma unroll
    for (int j = 0; j < 4; ++j) {
        float4 v = hr[j];
        s += fmaxf(v.x, 0.0f) * w[j * 4 + 0];
        s += fmaxf(v.y, 0.0f) * w[j * 4 + 1];
        s += fmaxf(v.z, 0.0f) * w[j * 4 + 2];
        s += fmaxf(v.w, 0.0f) * w[j * 4 + 3];
    }
    out[i] = s;
}

// out[i*32+c] *= log(exp(t[i]) + 1.1)
__global__ void k_finalize(const float* __restrict__ t,
                           float* __restrict__ out, int n) {
    int i = gtid();
    if (i >= n * 32) return;
    float tv = logf(expf(t[i >> 5]) + 1.1f);
    out[i] *= tv;
}

// ---------------------------------------------------------------------------

extern "C" void kernel_launch(void* const* d_in, const int* in_sizes, int n_in,
                              void* d_out, int out_size, void* d_ws, size_t ws_size,
                              hipStream_t stream) {
    const float* x      = (const float*)d_in[0];   // [N,128]
    const int*   ei     = (const int*)  d_in[1];   // [2,E]
    const float* W_base = (const float*)d_in[2];   // [128,32]
    const float* b_base = (const float*)d_in[3];   // [32]
    const float* W1     = (const float*)d_in[4];   // [32,16]
    const float* b1     = (const float*)d_in[5];   // [16]
    const float* W2     = (const float*)d_in[6];   // [16,1]
    const float* b2     = (const float*)d_in[7];   // [1]
    float* out = (float*)d_out;                    // [N,32] (logits, then *t)

    const int N = in_sizes[0] / 128;
    const int E = in_sizes[1] / 2;
    const int* src = ei;
    const int* dst = ei + E;

    // workspace layout (floats; base is >=256B aligned, all offsets keep
    // 32B alignment for the v16h fragment loads)
    float* ws   = (float*)d_ws;
    float* deg  = ws;                       // N
    float* dinv = deg  + N;                 // N
    float* xw   = dinv + N;                 // N*32 : x @ W_base
    float* lw   = xw   + (size_t)N * 32;    // N*16 : logits @ W1
    float* h1   = lw   + (size_t)N * 16;    // N*16 : pre-relu agg2
    float* tlin = h1   + (size_t)N * 16;    // N    : relu(h1) @ W2
    float* tbuf = tlin + N;                 // N    : aggregated temperature
    _Float16* bp1 = (_Float16*)(tbuf + N);  // 4 chunks * 2 tiles * 512 halves
    _Float16* bp2 = bp1 + 4 * 2 * 512;      // 1 chunk  * 1 tile  * 512 halves

    const int thr = 256;
    auto cdiv = [](int a, int b) { return (a + b - 1) / b; };
    const int mtiles = cdiv(N, 16);
    const int wpb = thr / 32;               // waves per block

    // 1) symmetric normalization (+ pack both weight matrices, independent)
    k_init_deg <<<cdiv(N, thr), thr, 0, stream>>>(deg, N);
    k_accum_deg<<<cdiv(E, thr), thr, 0, stream>>>(dst, E, deg);
    k_dinv     <<<cdiv(N, thr), thr, 0, stream>>>(deg, dinv, N);
    k_pack_b<2><<<1, 256, 0, stream>>>(W_base, bp1, 128);
    k_pack_b<1><<<1,  32, 0, stream>>>(W1,     bp2, 32);

    // 2) layer 1: logits = agg(x @ W_base) + b_base   -> written into d_out
    k_wmma_gemm<2><<<cdiv(mtiles, wpb), thr, 0, stream>>>(x, bp1, xw, N, 128);
    k_init_agg<32><<<cdiv(N * 32, thr), thr, 0, stream>>>(xw, dinv, b_base, out, N);
    k_scatter<32> <<<cdiv(E * 32, thr), thr, 0, stream>>>(xw, src, dst, dinv, out, E);

    // 3) layer 2: h1 = agg(logits @ W1) + b1   (ReLU deferred to matvec read)
    k_wmma_gemm<1><<<cdiv(mtiles, wpb), thr, 0, stream>>>(out, bp2, lw, N, 32);
    k_init_agg<16><<<cdiv(N * 16, thr), thr, 0, stream>>>(lw, dinv, b1, h1, N);
    k_scatter<16> <<<cdiv(E * 16, thr), thr, 0, stream>>>(lw, src, dst, dinv, h1, E);

    // 4) layer 3: t = softplus-ish(agg(relu(h1) @ W2) + b2)
    k_matvec16_relu<<<cdiv(N, thr), thr, 0, stream>>>(h1, W2, tlin, N);
    k_init_agg<1><<<cdiv(N, thr), thr, 0, stream>>>(tlin, dinv, b2, tbuf, N);
    k_scatter<1> <<<cdiv(E, thr), thr, 0, stream>>>(tlin, src, dst, dinv, tbuf, E);

    // 5) out = logits * t
    k_finalize <<<cdiv(N * 32, thr), thr, 0, stream>>>(tbuf, out, N);
}
